// SignedAttention_38165079392508
// MI455X (gfx1250) — matrix operand order repaired and checked by reading, not compiled, not run on hardware
//
#include <hip/hip_runtime.h>

typedef __attribute__((ext_vector_type(16))) __bf16 v16bf;
typedef __attribute__((ext_vector_type(8)))  float  v8f;

#define D_DIM 64
#define H_DIM 4
#define HD    256   // H_DIM * D_DIM

// ---- ordered-uint encoding of float for atomicMax-based segment max ----
__device__ __forceinline__ unsigned enc_ord(float f) {
    unsigned u = __float_as_uint(f);
    return (u & 0x80000000u) ? ~u : (u | 0x80000000u);
}
__device__ __forceinline__ float dec_ord(unsigned u) {
    return (u & 0x80000000u) ? __uint_as_float(u & 0x7FFFFFFFu)
                             : __uint_as_float(~u);
}

// ---- init workspace: accum = 0, ssum = 0, mmax = enc(-inf) ----
__global__ void init_ws(float* __restrict__ accum, float* __restrict__ ssum,
                        unsigned* __restrict__ mmax, int N) {
    int idx = blockIdx.x * blockDim.x + threadIdx.x;
    if (idx < N * HD) accum[idx] = 0.0f;
    if (idx < N * H_DIM) {
        ssum[idx] = 0.0f;
        mmax[idx] = 0x007FFFFFu;   // enc_ord(-inf)
    }
}

// =====================================================================
// Fragment packing, contiguous k order used by this code (not AMD's table order). bf16 A-fragment (16x32):
//   lane<16: row=lane,    elems 0-15 -> K 0-15
//   lane>=16: row=lane-16, elems 0-15 -> K 16-31
// B-fragment (32x16): lane<16: col=lane, K 0-15; lane>=16: col=lane-16, K 16-31.
// Packed layout: tile-major, 32 lanes x 16 bf16 contiguous per (tile, kblock).
// =====================================================================

// pack A-side matrix (rows x 32*KBT f32, row-major, row pitch = 32*KBT)
// into fragment order. One thread per (mtile, kblock, lane).
template <int KBT>
__global__ void pack_A(const float* __restrict__ A, __bf16* __restrict__ Ap, int total) {
    int tid = blockIdx.x * blockDim.x + threadIdx.x;
    if (tid >= total) return;
    int lane = tid & 31;
    int kb   = (tid >> 5) % KBT;
    int mt   = tid / (32 * KBT);
    int row  = mt * 16 + (lane & 15);
    int ksel = (lane < 16) ? 0 : 16;
    const float* ar = A + (size_t)row * (32 * KBT) + kb * 32;
    float4 a0 = *(const float4*)(ar + ksel);
    float4 a1 = *(const float4*)(ar + ksel + 4);
    float4 a2 = *(const float4*)(ar + ksel + 8);
    float4 a3 = *(const float4*)(ar + ksel + 12);
    v16bf o;
    o[0]  = (__bf16)a0.x; o[1]  = (__bf16)a0.y; o[2]  = (__bf16)a0.z; o[3]  = (__bf16)a0.w;
    o[4]  = (__bf16)a1.x; o[5]  = (__bf16)a1.y; o[6]  = (__bf16)a1.z; o[7]  = (__bf16)a1.w;
    o[8]  = (__bf16)a2.x; o[9]  = (__bf16)a2.y; o[10] = (__bf16)a2.z; o[11] = (__bf16)a2.w;
    o[12] = (__bf16)a3.x; o[13] = (__bf16)a3.y; o[14] = (__bf16)a3.z; o[15] = (__bf16)a3.w;
    *(v16bf*)(Ap + (size_t)tid * 16) = o;
}

// pack B-side weight (Kt x Ncols f32, row-major) into fragment order.
// One thread per (ntile, kblock, lane); tid = (nt*KBT + kb)*32 + lane.
__global__ void pack_B(const float* __restrict__ W, __bf16* __restrict__ Wp,
                       int KBT, int Ncols, int total) {
    int tid = blockIdx.x * blockDim.x + threadIdx.x;
    if (tid >= total) return;
    int lane = tid & 31;
    int kb   = (tid >> 5) % KBT;
    int nt   = tid / (32 * KBT);
    int col  = nt * 16 + (lane & 15);
    int kc   = (lane < 16) ? 0 : 16;
    v16bf o;
    #pragma unroll
    for (int i = 0; i < 16; ++i)
        o[i] = (__bf16)W[(size_t)(kb * 32 + kc + i) * Ncols + col];
    *(v16bf*)(Wp + (size_t)tid * 16) = o;
}

// ---- QKV projection: wave per 16x16 tile; packed fragments; 6 WMMAs ----
__global__ void __launch_bounds__(32)
qkv_proj(const __bf16* __restrict__ Xp,
         const __bf16* __restrict__ Wqp, const float* __restrict__ bq,
         const __bf16* __restrict__ Wkp, const float* __restrict__ bk,
         const __bf16* __restrict__ Wvp, const float* __restrict__ bv,
         float* __restrict__ Qb, float* __restrict__ Kb, float* __restrict__ Vb) {
    const int lane = threadIdx.x;
    const int mt = blockIdx.y;
    const int nt = blockIdx.x;            // 0..15
    const int col = nt * 16 + (lane & 15);

    v8f aq = {}, ak = {}, av = {};
    #pragma unroll
    for (int kb = 0; kb < 2; ++kb) {
        v16bf af = *(const v16bf*)(Xp  + ((size_t)(mt * 2 + kb) * 32 + lane) * 16);
        v16bf fq = *(const v16bf*)(Wqp + ((size_t)(nt * 2 + kb) * 32 + lane) * 16);
        v16bf fk = *(const v16bf*)(Wkp + ((size_t)(nt * 2 + kb) * 32 + lane) * 16);
        v16bf fv = *(const v16bf*)(Wvp + ((size_t)(nt * 2 + kb) * 32 + lane) * 16);
        aq = __builtin_amdgcn_wmma_f32_16x16x32_bf16(false, af, false, fq, (short)0, aq, false, false);
        ak = __builtin_amdgcn_wmma_f32_16x16x32_bf16(false, af, false, fk, (short)0, ak, false, false);
        av = __builtin_amdgcn_wmma_f32_16x16x32_bf16(false, af, false, fv, (short)0, av, false, false);
    }
    const int rb = (lane < 16) ? 0 : 8;
    #pragma unroll
    for (int r = 0; r < 8; ++r) {
        size_t o = (size_t)(mt * 16 + rb + r) * HD + col;
        Qb[o] = aq[r] + bq[col];
        Kb[o] = ak[r] + bk[col];
        Vb[o] = av[r] + bv[col];
    }
}

// ---- pass 1: per-edge scores + segment max (wave per edge) ----
__global__ void __launch_bounds__(256)
edge_scores(const float* __restrict__ Qb, const float* __restrict__ Kb,
            const int* __restrict__ src, const int* __restrict__ tgt,
            const float* __restrict__ sign,
            float* __restrict__ scoreBuf, unsigned* __restrict__ mmax, int E) {
    int wid  = blockIdx.x * 8 + (threadIdx.x >> 5);
    int lane = threadIdx.x & 31;
    if (wid >= E) return;
    int s = src[wid], t = tgt[wid];
    const float4* Qp = (const float4*)(Qb + (size_t)s * HD);
    const float4* Kp = (const float4*)(Kb + (size_t)t * HD);
    float4 q0 = Qp[lane * 2], q1 = Qp[lane * 2 + 1];
    float4 k0 = Kp[lane * 2], k1 = Kp[lane * 2 + 1];
    float p = q0.x * k0.x + q0.y * k0.y + q0.z * k0.z + q0.w * k0.w
            + q1.x * k1.x + q1.y * k1.y + q1.z * k1.z + q1.w * k1.w;
    p += __shfl_xor(p, 1, 32);
    p += __shfl_xor(p, 2, 32);
    p += __shfl_xor(p, 4, 32);
    if ((lane & 7) == 0) {
        int h = lane >> 3;
        float score = p * 0.125f * sign[s];   // 1/sqrt(64)
        scoreBuf[(size_t)wid * H_DIM + h] = score;
        atomicMax(&mmax[s * H_DIM + h], enc_ord(score));
    }
}

// ---- pass 2: ex = exp(score - m); segment sum ----
__global__ void __launch_bounds__(256)
edge_exp(const int* __restrict__ src, float* __restrict__ scoreBuf,
         const unsigned* __restrict__ mmax, float* __restrict__ ssum, int EH) {
    int idx = blockIdx.x * blockDim.x + threadIdx.x;
    if (idx >= EH) return;
    int e = idx >> 2, h = idx & 3;
    int s = src[e];
    float m  = dec_ord(mmax[s * H_DIM + h]);
    float ex = __expf(scoreBuf[idx] - m);
    scoreBuf[idx] = ex;
    unsafeAtomicAdd(&ssum[s * H_DIM + h], ex);
}

// ---- pass 3: weighted scatter of V[tgt] into accum[src] (wave per edge) ----
__global__ void __launch_bounds__(256)
edge_scatter(const float* __restrict__ Vb,
             const int* __restrict__ src, const int* __restrict__ tgt,
             const float* __restrict__ exBuf, const float* __restrict__ ssum,
             float* __restrict__ accum, int E) {
    int wid  = blockIdx.x * 8 + (threadIdx.x >> 5);
    int lane = threadIdx.x & 31;
    if (wid >= E) return;
    int s = src[wid], t = tgt[wid];
    int h = lane >> 3;
    float attn = exBuf[(size_t)wid * H_DIM + h] / (ssum[s * H_DIM + h] + 1e-10f);
    const float4* Vp = (const float4*)(Vb + (size_t)t * HD);
    float4 v0 = Vp[lane * 2], v1 = Vp[lane * 2 + 1];
    float* dst = accum + (size_t)s * HD + lane * 8;
    unsafeAtomicAdd(dst + 0, v0.x * attn);
    unsafeAtomicAdd(dst + 1, v0.y * attn);
    unsafeAtomicAdd(dst + 2, v0.z * attn);
    unsafeAtomicAdd(dst + 3, v0.w * attn);
    unsafeAtomicAdd(dst + 4, v1.x * attn);
    unsafeAtomicAdd(dst + 5, v1.y * attn);
    unsafeAtomicAdd(dst + 6, v1.z * attn);
    unsafeAtomicAdd(dst + 7, v1.w * attn);
}

// ---- output projection: packed accum [N,256] @ packed Wo [256,64] + bo ----
__global__ void __launch_bounds__(32)
out_proj(const __bf16* __restrict__ Ap, const __bf16* __restrict__ Wop,
         const float* __restrict__ bo, float* __restrict__ out) {
    const int lane = threadIdx.x;
    const int mt = blockIdx.y;
    const int nt = blockIdx.x;            // 0..3
    const int col = nt * 16 + (lane & 15);

    v8f acc = {};
    #pragma unroll
    for (int kb = 0; kb < 8; ++kb) {
        v16bf af = *(const v16bf*)(Ap  + ((size_t)(mt * 8 + kb) * 32 + lane) * 16);
        v16bf bf = *(const v16bf*)(Wop + ((size_t)(nt * 8 + kb) * 32 + lane) * 16);
        acc = __builtin_amdgcn_wmma_f32_16x16x32_bf16(false, af, false, bf, (short)0, acc, false, false);
    }
    const int rb = (lane < 16) ? 0 : 8;
    #pragma unroll
    for (int r = 0; r < 8; ++r) {
        out[(size_t)(mt * 16 + rb + r) * D_DIM + col] = acc[r] + bo[col];
    }
}

extern "C" void kernel_launch(void* const* d_in, const int* in_sizes, int n_in,
                              void* d_out, int out_size, void* d_ws, size_t ws_size,
                              hipStream_t stream) {
    const float* X    = (const float*)d_in[0];   // [N,64]
    const float* sign = (const float*)d_in[1];   // [N]
    const int*   src  = (const int*)  d_in[2];   // [E]
    const int*   tgt  = (const int*)  d_in[3];   // [E]
    const float* Wq   = (const float*)d_in[4];
    const float* bq   = (const float*)d_in[5];
    const float* Wk   = (const float*)d_in[6];
    const float* bk   = (const float*)d_in[7];
    const float* Wv   = (const float*)d_in[8];
    const float* bv   = (const float*)d_in[9];
    const float* Wo   = (const float*)d_in[10];
    const float* bo   = (const float*)d_in[11];

    const int N = in_sizes[0] / D_DIM;   // 50000 (divisible by 16)
    const int E = in_sizes[2];           // 800000
    const int MT = N / 16;               // 3125 row tiles

    // ---- workspace carve-up (all chunks 256B-aligned by construction) ----
    char* ws = (char*)d_ws;
    const size_t szNHD = (size_t)N * HD * sizeof(float);           // 51.2 MB
    const size_t szEH  = (size_t)E * H_DIM * sizeof(float);        // 12.8 MB
    const size_t szNH4 = (size_t)N * H_DIM * sizeof(unsigned);     // 0.8 MB
    float*    Qb       = (float*)(ws);
    float*    Kb       = (float*)(ws + szNHD);
    float*    Vb       = (float*)(ws + 2 * szNHD);
    float*    accum    = (float*)(ws + 3 * szNHD);
    float*    scoreBuf = (float*)(ws + 4 * szNHD);
    unsigned* mmax     = (unsigned*)(ws + 4 * szNHD + szEH);
    float*    ssum     = (float*)(ws + 4 * szNHD + szEH + szNH4);
    char*     p        = ws + 4 * szNHD + szEH + 2 * szNH4;
    __bf16*   Xp  = (__bf16*)p;  p += (size_t)N * D_DIM * sizeof(__bf16);   // 6.4 MB
    __bf16*   Wqp = (__bf16*)p;  p += (size_t)D_DIM * HD * sizeof(__bf16);
    __bf16*   Wkp = (__bf16*)p;  p += (size_t)D_DIM * HD * sizeof(__bf16);
    __bf16*   Wvp = (__bf16*)p;  p += (size_t)D_DIM * HD * sizeof(__bf16);
    __bf16*   Wop = (__bf16*)p;  p += (size_t)HD * D_DIM * sizeof(__bf16);
    __bf16*   Ap  = (__bf16*)p;  /* N*HD bf16 = 25.6 MB */

    // 1) init accumulators
    init_ws<<<(N * HD + 255) / 256, 256, 0, stream>>>(accum, ssum, mmax, N);

    // 2) pack GEMM operands to fragment-ordered bf16
    {
        int totX = MT * 2 * 32;                       // KBT=2 for X (K=64)
        pack_A<2><<<(totX + 255) / 256, 256, 0, stream>>>(X, Xp, totX);
        int totW = 16 * 2 * 32;                       // 16 ntiles, KBT=2
        pack_B<<<(totW + 255) / 256, 256, 0, stream>>>(Wq, Wqp, 2, HD, totW);
        pack_B<<<(totW + 255) / 256, 256, 0, stream>>>(Wk, Wkp, 2, HD, totW);
        pack_B<<<(totW + 255) / 256, 256, 0, stream>>>(Wv, Wvp, 2, HD, totW);
        int totWo = 4 * 8 * 32;                       // 4 ntiles, KBT=8
        pack_B<<<(totWo + 255) / 256, 256, 0, stream>>>(Wo, Wop, 8, D_DIM, totWo);
    }

    // 3) QKV projection (WMMA)
    {
        dim3 g(HD / 16, MT);
        qkv_proj<<<g, 32, 0, stream>>>(Xp, Wqp, bq, Wkp, bk, Wvp, bv, Qb, Kb, Vb);
    }
    // 4) scores + segment max
    edge_scores<<<(E + 7) / 8, 256, 0, stream>>>(Qb, Kb, src, tgt, sign, scoreBuf, mmax, E);
    // 5) exp + segment sum
    edge_exp<<<(E * H_DIM + 255) / 256, 256, 0, stream>>>(src, scoreBuf, mmax, ssum, E * H_DIM);
    // 6) weighted scatter
    edge_scatter<<<(E + 7) / 8, 256, 0, stream>>>(Vb, src, tgt, scoreBuf, ssum, accum, E);
    // 7) pack accumulator to fragment-ordered bf16 (KBT=8 for K=256)
    {
        int totA = MT * 8 * 32;
        pack_A<8><<<(totA + 255) / 256, 256, 0, stream>>>(accum, Ap, totA);
    }
    // 8) output projection (WMMA)
    {
        dim3 g(D_DIM / 16, MT);
        out_proj<<<g, 32, 0, stream>>>(Ap, Wop, bo, (float*)d_out);
    }
}
